// Attention_20925080666453
// MI455X (gfx1250) — matrix operand order repaired and checked by reading, not compiled, not run on hardware
//
#include <hip/hip_runtime.h>

typedef __attribute__((ext_vector_type(16))) _Float16 v16h;
typedef __attribute__((ext_vector_type(8)))  _Float16 v8h;
typedef __attribute__((ext_vector_type(8)))  float    v8f;
typedef __attribute__((ext_vector_type(4)))  float    v4f;

#define WMMA_F16(A, B, C) \
  __builtin_amdgcn_wmma_f32_16x16x32_f16(false, (A), false, (B), (short)0, (C), false, false)

static constexpr int Bsz = 2, Hn = 8, Sn = 4096, Dh = 64;
static constexpr int NEL  = Bsz * Hn * Sn * Dh;    // 4,194,304 per tensor
static constexpr int SBLK = 32;                    // query rows per block (2 WMMA row-sets)
static constexpr int EPITCH = SBLK * 32 + 8;       // f32 per head plane in eLDS (+8: bank skew)
static constexpr int AROW   = 40;                  // f16 per s-row in aLDS (32 + 8 pad, 16B aligned)
static constexpr int APITCH = SBLK * AROW;         // f16 per head plane

// ---- pass 1: f32 -> f16 (+ fold 1/sqrt(512) into Q; transpose K,V so every WMMA
//      B-fragment is a contiguous 32B load: Kt = [bh][t][d], Vt = [bh][d][t]) ----

__global__ void cvt_q_scale(const float* __restrict__ Q, _Float16* __restrict__ Q16, float scale) {
  int i = blockIdx.x * blockDim.x + threadIdx.x;
  Q16[i] = (_Float16)(Q[i] * scale);
}

__global__ void cvt_k_t(const float* __restrict__ K, _Float16* __restrict__ Kt) {
  // K: [bh][d][t] (t fastest, coalesced read) -> Kt: [bh][t][d]
  int i = blockIdx.x * blockDim.x + threadIdx.x;
  int t = i & (Sn - 1);
  int d = (i >> 12) & (Dh - 1);
  int bh = i >> 18;
  Kt[((size_t)bh * Sn + t) * Dh + d] = (_Float16)K[i];
}

__global__ void cvt_v_t(const float* __restrict__ V, _Float16* __restrict__ Vt) {
  // V: [bh][t][d] (d fastest, coalesced read) -> Vt: [bh][d][t]
  int i = blockIdx.x * blockDim.x + threadIdx.x;
  int d = i & (Dh - 1);
  int t = (i >> 6) & (Sn - 1);
  int bh = i >> 18;
  Vt[((size_t)bh * Dh + d) * Sn + t] = (_Float16)V[i];
}

// ---- pass 2: fused attention. One block = (b, 32-row s-block); wave w == head h.
//      softmax(axis=heads) is pointwise in (s,t): reduce across the 8 resident waves
//      through LDS per 32-wide t-tile, no streaming-softmax state needed. ----

__global__ __launch_bounds__(256) void attn_headsoftmax(
    const _Float16* __restrict__ Q16,  // [bh][s][d], pre-scaled
    const _Float16* __restrict__ Kt,   // [bh][t][d]
    const _Float16* __restrict__ Vt,   // [bh][d][t]
    float* __restrict__ out)           // [bh][s][d] (the torch .view is a raw reshape)
{
  __shared__ float    eLDS[Hn * EPITCH];   // e tiles, [h][t(0..31)][s(0..31)]
  __shared__ _Float16 aLDS[Hn * APITCH];   // softmaxed tiles, [h][s(0..31)][t(0..31)]

  const int tid  = threadIdx.x;
  const int lane = tid & 31;
  const int h    = tid >> 5;                       // wave index == head
  const int b    = blockIdx.x >> 7;                // 128 s-blocks per batch
  const int s0   = (blockIdx.x & 127) * SBLK;
  const int bh   = b * Hn + h;

  const int  lmod   = lane & 15;
  const bool lolane = lane < 16;
  const int  klo    = lolane ? 0 : 8;        // unused: A-fragments also take kcol
  const int  kcol   = lolane ? 0 : 16;       // B-fragment k sub-offset (halves)
  const int  srow   = lolane ? 0 : 8;        // C/D-fragment row sub-offset

  // Q A-fragments: 2 row-sets x (d 0..31, d 32..63)
  v16h aQ0[2], aQ1[2];
#pragma unroll
  for (int r = 0; r < 2; ++r) {
    const _Float16* qrow = Q16 + ((size_t)bh * Sn + s0 + r * 16 + lmod) * Dh;
    v8h q0l = *(const v8h*)(qrow + kcol);
    v8h q0h = *(const v8h*)(qrow + 8 + kcol);
    v8h q1l = *(const v8h*)(qrow + 32 + kcol);
    v8h q1h = *(const v8h*)(qrow + 40 + kcol);
    aQ0[r] = __builtin_shufflevector(q0l, q0h, 0,1,2,3,4,5,6,7,8,9,10,11,12,13,14,15);
    aQ1[r] = __builtin_shufflevector(q1l, q1h, 0,1,2,3,4,5,6,7,8,9,10,11,12,13,14,15);
  }

  v8f oacc[2][4];
#pragma unroll
  for (int r = 0; r < 2; ++r)
#pragma unroll
    for (int dn = 0; dn < 4; ++dn) oacc[r][dn] = v8f{};

  const _Float16* kbase = Kt + (size_t)bh * Sn * Dh;
  const _Float16* vbase = Vt + (size_t)bh * Dh * Sn;

  for (int t0 = 0; t0 < Sn; t0 += 32) {
    // ---- batch ALL fragment loads for this t-tile up front: the QK stage only has
    //      to wait for the K half, and the V half completes behind QK+softmax. ----
    v16h bK0[2], bK1[2], bV[4];
#pragma unroll
    for (int ts = 0; ts < 2; ++ts) {
      const _Float16* krow = kbase + (size_t)(t0 + ts * 16 + lmod) * Dh;
      bK0[ts] = *(const v16h*)(krow + kcol);        // d 0..31
      bK1[ts] = *(const v16h*)(krow + 32 + kcol);   // d 32..63
    }
#pragma unroll
    for (int dn = 0; dn < 4; ++dn) {
      const _Float16* vrow = vbase + (size_t)(dn * 16 + lmod) * Sn + t0 + kcol;
      bV[dn] = *(const v16h*)(vrow);
    }

    // ---- warm caches for the NEXT t-tile (global_prefetch_b8: no VGPR, no LOADcnt) ----
    if (t0 + 32 < Sn) {
#pragma unroll
      for (int ts = 0; ts < 2; ++ts)
        __builtin_prefetch(kbase + (size_t)(t0 + 32 + ts * 16 + lmod) * Dh, 0, 3);
#pragma unroll
      for (int dn = 0; dn < 4; ++dn)
        __builtin_prefetch(vbase + (size_t)(dn * 16 + lmod) * Sn + t0 + 32 + kcol, 0, 3);
    }

    // ---- QK^T: 32s x 32t scores; K fragments amortized over both row-sets ----
#pragma unroll
    for (int ts = 0; ts < 2; ++ts) {
#pragma unroll
      for (int r = 0; r < 2; ++r) {
        v8f e = {};
        e = WMMA_F16(aQ0[r], bK0[ts], e);
        e = WMMA_F16(aQ1[r], bK1[ts], e);
        float* ep = eLDS + h * EPITCH + (ts * 16 + lmod) * SBLK + r * 16 + srow;
        *(v4f*)(ep)     = __builtin_shufflevector(e, e, 0, 1, 2, 3);
        *(v4f*)(ep + 4) = __builtin_shufflevector(e, e, 4, 5, 6, 7);
      }
    }
    __syncthreads();

    // ---- softmax over HEADS: pointwise in (s,t); fixed 4 passes, no exec masking ----
#pragma unroll
    for (int it = 0; it < (32 * SBLK) / 256; ++it) {
      int p = tid + it * 256;
      int t = p >> 5, s = p & (SBLK - 1);
      float x[Hn];
      float m = -3.0e38f;
#pragma unroll
      for (int hh = 0; hh < Hn; ++hh) {
        x[hh] = eLDS[hh * EPITCH + t * SBLK + s];
        m = fmaxf(m, x[hh]);
      }
      float sum = 0.f;
#pragma unroll
      for (int hh = 0; hh < Hn; ++hh) { x[hh] = __expf(x[hh] - m); sum += x[hh]; }
      float inv = 1.0f / sum;
#pragma unroll
      for (int hh = 0; hh < Hn; ++hh)
        aLDS[hh * APITCH + s * AROW + t] = (_Float16)(x[hh] * inv);
    }
    __syncthreads();

    // ---- A-fragments of normalized scores for this head (both row-sets) ----
    v16h aA[2];
#pragma unroll
    for (int r = 0; r < 2; ++r) {
      const _Float16* arow = aLDS + h * APITCH + (r * 16 + lmod) * AROW;
      v8h al = *(const v8h*)(arow + kcol);
      v8h ah = *(const v8h*)(arow + 8 + kcol);
      aA[r] = __builtin_shufflevector(al, ah, 0,1,2,3,4,5,6,7,8,9,10,11,12,13,14,15);
    }

    // ---- A @ V: V fragments already resident ----
#pragma unroll
    for (int dn = 0; dn < 4; ++dn)
#pragma unroll
      for (int r = 0; r < 2; ++r) oacc[r][dn] = WMMA_F16(aA[r], bV[dn], oacc[r][dn]);
  }

  // ---- store: out is the flat [bh][s][d] buffer ----
#pragma unroll
  for (int r = 0; r < 2; ++r)
#pragma unroll
    for (int j = 0; j < 8; ++j) {
      float* po = out + ((size_t)bh * Sn + s0 + r * 16 + srow + j) * Dh + lmod;
      po[0]  = oacc[r][0][j];
      po[16] = oacc[r][1][j];
      po[32] = oacc[r][2][j];
      po[48] = oacc[r][3][j];
    }
}

extern "C" void kernel_launch(void* const* d_in, const int* in_sizes, int n_in,
                              void* d_out, int out_size, void* d_ws, size_t ws_size,
                              hipStream_t stream) {
  (void)in_sizes; (void)n_in; (void)out_size; (void)ws_size;
  const float* Q = (const float*)d_in[0];
  const float* K = (const float*)d_in[1];
  const float* V = (const float*)d_in[2];
  float* out = (float*)d_out;

  _Float16* Q16 = (_Float16*)d_ws;
  _Float16* Kt  = Q16 + NEL;
  _Float16* Vt  = Kt + NEL;

  const float scale = 0.04419417382415922f;  // 1/sqrt(512)

  cvt_q_scale<<<NEL / 256, 256, 0, stream>>>(Q, Q16, scale);
  cvt_k_t    <<<NEL / 256, 256, 0, stream>>>(K, Kt);
  cvt_v_t    <<<NEL / 256, 256, 0, stream>>>(V, Vt);

  attn_headsoftmax<<<Bsz * (Sn / SBLK), 256, 0, stream>>>(Q16, Kt, Vt, out);
}
